// SchNet_44332652429578
// MI455X (gfx1250) — matrix operand order repaired and checked by reading, not compiled, not run on hardware
//
#include <hip/hip_runtime.h>

// ---------------------------------------------------------------------------
// SchNet forward for MI455X (gfx1250).
//  * All 128-wide GEMMs run on V_WMMA_F32_16X16X32_F16 (f16 in, f32 accum).
//  * Filter network + CFConv aggregation fused per atom: the 64x128 W tile
//    never touches HBM (would be 256MB/interaction otherwise).
//  * Weights pre-swizzled into per-lane WMMA B-fragment order -> one 32B
//    contiguous load per lane per fragment.
//  * A-tiles staged to LDS with GLOBAL_LOAD_ASYNC_TO_LDS_B128 (ASYNCcnt)
//    when the toolchain exposes the builtin; sync fallback otherwise.
// ---------------------------------------------------------------------------

#define B_   8
#define A_   1024
#define NB_  64
#define NAB_ 128
#define NF_  128
#define NG_  25
#define NI_  3

typedef __attribute__((ext_vector_type(16))) _Float16 v16h;
typedef __attribute__((ext_vector_type(8)))  _Float16 v8h;
typedef __attribute__((ext_vector_type(8)))  float    v8f;

#if defined(__gfx1250__) && \
    __has_builtin(__builtin_amdgcn_global_load_async_to_lds_b128) && \
    __has_builtin(__builtin_amdgcn_s_wait_asynccnt)
#define USE_ASYNC_LDS 1
#else
#define USE_ASYNC_LDS 0
#endif

#if USE_ASYNC_LDS
typedef __attribute__((ext_vector_type(4))) int i32x4;
typedef __attribute__((address_space(1))) i32x4 as1_i32x4;
typedef __attribute__((address_space(3))) i32x4 as3_i32x4;
// 16B global -> LDS async copy (tracked by ASYNCcnt, no VGPR round trip)
__device__ __forceinline__ void async_cp16(void* lds_dst, const void* g_src) {
  __builtin_amdgcn_global_load_async_to_lds_b128(
      (as1_i32x4*)(uintptr_t)g_src,
      (as3_i32x4*)(uint32_t)(uintptr_t)lds_dst, 0, 0);
}
__device__ __forceinline__ void async_wait0() {
  __builtin_amdgcn_s_wait_asynccnt(0);
}
#endif

__device__ __forceinline__ float ssp_f(float x) {
  // shifted softplus: log(1+exp(x)) - log(2), numerically stable
  return fmaxf(x, 0.0f) + log1pf(expf(-fabsf(x))) - 0.69314718055994531f;
}

// A-matrix fragment: 16x32 f16, row-major source with leading dim `ld` halves.
// Layout: lanes 0-15 -> M=lane, K {0..15}; lanes 16-31 -> M=lane-16,
// K {16..31}.  Loaded as two 16B vectors.
__device__ __forceinline__ v16h load_A16(const _Float16* p, int ld, int lane) {
  const int row = lane & 15, hi = lane >> 4;
  const _Float16* base = p + row * ld + hi * 16;
  v8h c0 = *(const v8h*)(base);
  v8h c1 = *(const v8h*)(base + 8);
  return __builtin_shufflevector(c0, c1, 0,1,2,3,4,5,6,7,8,9,10,11,12,13,14,15);
}

// B fragment from pre-swizzled weights: layout (ktile, ntile, lane, 16 halves),
// so each lane does one contiguous 32-byte load.
__device__ __forceinline__ v16h load_B_swz(const _Float16* frag, int lane) {
  return *(const v16h*)(frag + lane * 16);
}

__device__ __forceinline__ v8f wmma_f16(v16h a, v16h b, v8f c) {
  return __builtin_amdgcn_wmma_f32_16x16x32_f16(false, a, false, b,
                                                (short)0, c, false, false);
}

// ---------------------------------------------------------------------------
// Preparation kernels
// ---------------------------------------------------------------------------

// Embedding gather: x = emb[z]  (f32 + f16 copies)
__global__ void emb_kernel(const float* __restrict__ emb,
                           const int* __restrict__ zn,
                           float* __restrict__ xF, _Float16* __restrict__ xH) {
  int idx = blockIdx.x * 256 + threadIdx.x;              // over B*A*128
  if (idx >= B_ * A_ * NAB_) return;
  int f = idx & 127, ba = idx >> 7;
  float v = emb[zn[ba] * NAB_ + f];
  xF[idx] = v;
  xH[idx] = (_Float16)v;
}

// Distances + cosine cutoff + gaussian smearing (K padded 25->32, f16).
// fij layout: (B*A, 64, 32) -> one atom's A-tile is a contiguous 4KB block.
__global__ void prep_kernel(const float* __restrict__ pos,
                            const float* __restrict__ cell,
                            const float* __restrict__ cofs,
                            const float* __restrict__ mask,
                            const int* __restrict__ nbh,
                            _Float16* __restrict__ fij,
                            float* __restrict__ CMout) {
  int idx = blockIdx.x * 256 + threadIdx.x;              // over B*A*NB
  if (idx >= B_ * A_ * NB_) return;
  int ba = idx >> 6;
  int b  = ba >> 10;
  int j  = nbh[idx];
  const float* pi = pos + (size_t)ba * 3;
  const float* pj = pos + ((size_t)b * A_ + j) * 3;
  const float* co = cofs + (size_t)idx * 3;
  const float* cl = cell + b * 9;
  // pos_j + cell_offset @ cell - pos_i   (einsum 'c,cd->d')
  float dx = pj[0] + co[0]*cl[0] + co[1]*cl[3] + co[2]*cl[6] - pi[0];
  float dy = pj[1] + co[0]*cl[1] + co[1]*cl[4] + co[2]*cl[7] - pi[1];
  float dz = pj[2] + co[0]*cl[2] + co[1]*cl[5] + co[2]*cl[8] - pi[2];
  float d2 = dx*dx + dy*dy + dz*dz;
  float mk = mask[idx];
  float r  = sqrtf(mk > 0.0f ? d2 : 1.0f) * mk;
  float C  = (r < 5.0f) ? 0.5f * (cosf(r * 0.62831853071795865f) + 1.0f) : 0.0f;
  CMout[idx] = C * mk;
  const float width = 5.0f / 24.0f;
  const float coeff = -0.5f / (width * width);
  __align__(16) _Float16 o[32];
#pragma unroll
  for (int g = 0; g < 32; ++g) {
    float t = r - (float)g * width;
    o[g] = (_Float16)((g < NG_) ? expf(coeff * t * t) : 0.0f);
  }
  uint4* d = (uint4*)(fij + (size_t)idx * 32);
  const uint4* s = (const uint4*)o;
  d[0] = s[0]; d[1] = s[1]; d[2] = s[2]; d[3] = s[3];
}

// Swizzle a (NI,128,128) f32 weight into WMMA B-fragment order:
// dst[mat][kt][nt][lane][j] = src[mat][kt*32 + (lane>>4)*16 + j][nt*16 + (lane&15)]
__global__ void swz128_kernel(const float* __restrict__ s,
                              _Float16* __restrict__ d) {
  int idx = blockIdx.x * 256 + threadIdx.x;   // over NI*4*8*32*16 = 49152
  if (idx >= NI_ * 4 * 8 * 32 * 16) return;
  int j   = idx & 15;
  int l   = (idx >> 4) & 31;
  int nt  = (idx >> 9) & 7;
  int kt  = (idx >> 12) & 3;
  int mat = idx >> 14;
  int krow = kt * 32 + (l >> 4) * 16 + j;
  int col  = nt * 16 + (l & 15);
  d[idx] = (_Float16)s[(size_t)mat * 16384 + (size_t)krow * 128 + col];
}

// Same for fw1 (NI,25,128): single k-tile, rows 25..31 zero-padded.
__global__ void swzfw1_kernel(const float* __restrict__ s,
                              _Float16* __restrict__ d) {
  int idx = blockIdx.x * 256 + threadIdx.x;   // over NI*8*32*16 = 12288
  if (idx >= NI_ * 8 * 32 * 16) return;
  int j   = idx & 15;
  int l   = (idx >> 4) & 31;
  int nt  = (idx >> 9) & 7;
  int mat = idx >> 12;
  int krow = (l >> 4) * 16 + j;
  int col  = nt * 16 + (l & 15);
  d[idx] = (krow < NG_)
               ? (_Float16)s[((size_t)mat * NG_ + krow) * 128 + col]
               : (_Float16)0.0f;
}

// ---------------------------------------------------------------------------
// 64-row x (K=128) x (N=128) WMMA GEMM with selectable epilogue.
//  EPI=0: outF = A@B (+bias)                        [y = x @ in2f]
//  EPI=1: outH = f16(ssp(A@B + bias))               [t = ssp(agg @ f2out + b)]
//  EPI=2: outF = Xin + A@B + bias; outH = f16(outF) [x += t @ dense + b]
// Block: 256 threads (8 waves), wave w owns output columns [16w,16w+16).
// Bw is pre-swizzled: (ktile, ntile, lane, 16) fragments.
// ---------------------------------------------------------------------------
template <int EPI>
__global__ __launch_bounds__(256, 2)
void gemm128_kernel(const _Float16* __restrict__ Amat,  // (rows,128) f16
                    const _Float16* __restrict__ Bw,    // swizzled (4,8,32,16)
                    const float* __restrict__ bias,     // (128) or null
                    const float* __restrict__ Xin,      // (rows,128) for EPI=2
                    float* __restrict__ outF,
                    _Float16* __restrict__ outH) {
  __shared__ __align__(16) _Float16 sA[64 * 128];
  const int rowBase = blockIdx.x * 64;
  {
    const uint4* src = (const uint4*)(Amat + (size_t)rowBase * 128);
    uint4* dst = (uint4*)sA;
#pragma unroll
    for (int t = 0; t < 4; ++t) {
#if USE_ASYNC_LDS
      async_cp16(&dst[threadIdx.x + 256 * t], &src[threadIdx.x + 256 * t]);
#else
      dst[threadIdx.x + 256 * t] = src[threadIdx.x + 256 * t];
#endif
    }
#if USE_ASYNC_LDS
    async_wait0();
#endif
  }
  __syncthreads();
  const int wave = threadIdx.x >> 5, lane = threadIdx.x & 31;
  const int col = lane & 15, hi = lane >> 4;
  const int colg = wave * 16 + col;
  const float bv = bias ? bias[colg] : 0.0f;
  v8f acc[4];
#pragma unroll
  for (int m = 0; m < 4; ++m)
#pragma unroll
    for (int r = 0; r < 8; ++r) acc[m][r] = bv;
#pragma unroll
  for (int k = 0; k < 4; ++k) {
    v16h b = load_B_swz(Bw + ((k * 8 + wave) * 32) * 16, lane);
#pragma unroll
    for (int m = 0; m < 4; ++m) {
      v16h a = load_A16(sA + m * 16 * 128 + k * 32, 128, lane);
      acc[m] = wmma_f16(a, b, acc[m]);
    }
  }
#pragma unroll
  for (int m = 0; m < 4; ++m)
#pragma unroll
    for (int r = 0; r < 8; ++r) {
      const int row = rowBase + m * 16 + hi * 8 + r;
      const size_t o = (size_t)row * 128 + colg;
      const float v = acc[m][r];
      if (EPI == 0) outF[o] = v;
      if (EPI == 1) outH[o] = (_Float16)ssp_f(v);
      if (EPI == 2) {
        float nv = Xin[o] + v;
        outF[o] = nv;
        outH[o] = (_Float16)nv;
      }
    }
}

// ---------------------------------------------------------------------------
// Fused CFConv: per atom (one block), H = ssp(fij@fw1+b1) (64x128, LDS, f16),
// W = H@fw2+b2 (registers), agg = sum_n W[n]*C[n]*mask[n]*y[nbh[n]].
// ---------------------------------------------------------------------------
__global__ __launch_bounds__(256, 2)
void cfconv_kernel(const _Float16* __restrict__ fij,  // (BA,64,32)
                   const float* __restrict__ CM,      // (BA,64)  C*mask
                   const int* __restrict__ nbh,       // (BA,64)
                   const float* __restrict__ y,       // (BA,128) f32
                   const _Float16* __restrict__ fw1,  // swizzled (1,8,32,16)
                   const float* __restrict__ fb1,     // (128)
                   const _Float16* __restrict__ fw2,  // swizzled (4,8,32,16)
                   const float* __restrict__ fb2,     // (128)
                   _Float16* __restrict__ aggH) {     // (BA,128) f16
  __shared__ __align__(16) _Float16 sF[64 * 32];
  __shared__ __align__(16) _Float16 sH[64 * 128];
  __shared__ float sCM[64];
  __shared__ int sIdx[64];
  const int ba = blockIdx.x;
  const int bBase = (ba >> 10) << 10;  // b * A
  {
    const uint4* src = (const uint4*)(fij + (size_t)ba * 64 * 32);
    uint4* dst = (uint4*)sF;
#if USE_ASYNC_LDS
    async_cp16(&dst[threadIdx.x], &src[threadIdx.x]);   // 256 x 16B = 4KB
#else
    dst[threadIdx.x] = src[threadIdx.x];
#endif
    if (threadIdx.x < 64) {
      sCM[threadIdx.x]  = CM[ba * 64 + threadIdx.x];
      sIdx[threadIdx.x] = nbh[ba * 64 + threadIdx.x];
    }
#if USE_ASYNC_LDS
    async_wait0();
#endif
  }
  __syncthreads();
  const int wave = threadIdx.x >> 5, lane = threadIdx.x & 31;
  const int col = lane & 15, hi = lane >> 4;
  const int colg = wave * 16 + col;
  // --- GEMM1 (K=32, one WMMA per M-tile) + ssp -> sH (f16) ---
  {
    const float bv = fb1[colg];
    v16h b = load_B_swz(fw1 + (wave * 32) * 16, lane);
#pragma unroll
    for (int m = 0; m < 4; ++m) {
      v8f c;
#pragma unroll
      for (int r = 0; r < 8; ++r) c[r] = bv;
      v16h a = load_A16(sF + m * 16 * 32, 32, lane);
      c = wmma_f16(a, b, c);
#pragma unroll
      for (int r = 0; r < 8; ++r)
        sH[(m * 16 + hi * 8 + r) * 128 + colg] = (_Float16)ssp_f(c[r]);
    }
  }
  __syncthreads();
  // --- GEMM2: W = H @ fw2 + fb2 (kept in registers) ---
  const float bv2 = fb2[colg];
  v8f acc[4];
#pragma unroll
  for (int m = 0; m < 4; ++m)
#pragma unroll
    for (int r = 0; r < 8; ++r) acc[m][r] = bv2;
#pragma unroll
  for (int k = 0; k < 4; ++k) {
    v16h b = load_B_swz(fw2 + ((k * 8 + wave) * 32) * 16, lane);
#pragma unroll
    for (int m = 0; m < 4; ++m) {
      v16h a = load_A16(sH + m * 16 * 128 + k * 32, 128, lane);
      acc[m] = wmma_f16(a, b, acc[m]);
    }
  }
  // --- modulate by C*mask, gather y over neighbors, reduce 64 rows ---
  float part = 0.0f;
#pragma unroll
  for (int m = 0; m < 4; ++m)
#pragma unroll
    for (int r = 0; r < 8; ++r) {
      const int row = m * 16 + hi * 8 + r;
      const float yv = y[(size_t)(bBase + sIdx[row]) * 128 + colg];
      part = fmaf(acc[m][r] * sCM[row], yv, part);
    }
  part += __shfl_xor(part, 16, 32);  // combine the two lane halves (rows +8)
  if (hi == 0) aggH[(size_t)ba * 128 + colg] = (_Float16)part;
}

// ---------------------------------------------------------------------------
// Host-side orchestration
// ---------------------------------------------------------------------------
extern "C" void kernel_launch(void* const* d_in, const int* in_sizes, int n_in,
                              void* d_out, int out_size, void* d_ws,
                              size_t ws_size, hipStream_t stream) {
  (void)in_sizes; (void)n_in; (void)out_size; (void)ws_size;
  const float* positions = (const float*)d_in[0];
  const float* cell      = (const float*)d_in[1];
  const float* cofs      = (const float*)d_in[2];
  const float* nmask     = (const float*)d_in[3];
  /* d_in[4] atom_mask unused by reference math */
  const float* emb       = (const float*)d_in[5];
  const float* fw1       = (const float*)d_in[6];
  const float* fb1       = (const float*)d_in[7];
  const float* fw2       = (const float*)d_in[8];
  const float* fb2       = (const float*)d_in[9];
  const float* in2f_w    = (const float*)d_in[10];
  const float* f2out_w   = (const float*)d_in[11];
  const float* f2out_b   = (const float*)d_in[12];
  const float* dense_w   = (const float*)d_in[13];
  const float* dense_b   = (const float*)d_in[14];
  const int*   zn        = (const int*)d_in[15];
  const int*   nbh       = (const int*)d_in[16];

  const int BA = B_ * A_;
  char* ws = (char*)d_ws;
  size_t off = 0;
  auto carve = [&](size_t bytes) -> char* {
    char* p = ws + off;
    off += (bytes + 255) & ~(size_t)255;
    return p;
  };
  float*     xF    = (float*)carve((size_t)BA * 128 * 4);
  float*     yF    = (float*)carve((size_t)BA * 128 * 4);
  float*     CMf   = (float*)carve((size_t)BA * 64 * 4);
  _Float16*  xH    = (_Float16*)carve((size_t)BA * 128 * 2);
  _Float16*  aggH  = (_Float16*)carve((size_t)BA * 128 * 2);
  _Float16*  tH    = (_Float16*)carve((size_t)BA * 128 * 2);
  _Float16*  fijH  = (_Float16*)carve((size_t)BA * 64 * 32 * 2);
  _Float16*  fw1H  = (_Float16*)carve((size_t)NI_ * 8 * 32 * 16 * 2);
  _Float16*  fw2H  = (_Float16*)carve((size_t)NI_ * 4 * 8 * 32 * 16 * 2);
  _Float16*  in2fH = (_Float16*)carve((size_t)NI_ * 4 * 8 * 32 * 16 * 2);
  _Float16*  f2oH  = (_Float16*)carve((size_t)NI_ * 4 * 8 * 32 * 16 * 2);
  _Float16*  denH  = (_Float16*)carve((size_t)NI_ * 4 * 8 * 32 * 16 * 2);

  const int NW = NI_ * 128 * 128;  // 49152 elems per big weight family
  // weight swizzle into WMMA B-fragment order (f32 -> f16)
  swzfw1_kernel<<<(NI_ * 8 * 32 * 16 + 255) / 256, 256, 0, stream>>>(fw1, fw1H);
  swz128_kernel<<<(NW + 255) / 256, 256, 0, stream>>>(fw2,     fw2H);
  swz128_kernel<<<(NW + 255) / 256, 256, 0, stream>>>(in2f_w,  in2fH);
  swz128_kernel<<<(NW + 255) / 256, 256, 0, stream>>>(f2out_w, f2oH);
  swz128_kernel<<<(NW + 255) / 256, 256, 0, stream>>>(dense_w, denH);
  // embedding + gaussian basis
  emb_kernel<<<(BA * 128 + 255) / 256, 256, 0, stream>>>(emb, zn, xF, xH);
  prep_kernel<<<(BA * 64 + 255) / 256, 256, 0, stream>>>(
      positions, cell, cofs, nmask, nbh, fijH, CMf);

  const int nRowTiles = BA / 64;  // 128 blocks per 64-row GEMM
  for (int i = 0; i < NI_; ++i) {
    const _Float16* w_in2f = in2fH + (size_t)i * 16384;
    const _Float16* w_fw1  = fw1H  + (size_t)i * 4096;
    const _Float16* w_fw2  = fw2H  + (size_t)i * 16384;
    const _Float16* w_f2o  = f2oH  + (size_t)i * 16384;
    const _Float16* w_den  = denH  + (size_t)i * 16384;
    // y = x @ in2f_w[i]
    gemm128_kernel<0><<<nRowTiles, 256, 0, stream>>>(
        xH, w_in2f, nullptr, nullptr, yF, nullptr);
    // fused filter network + CFConv aggregation
    cfconv_kernel<<<BA, 256, 0, stream>>>(
        fijH, CMf, nbh, yF, w_fw1, fb1 + i * 128, w_fw2, fb2 + i * 128, aggH);
    // t = ssp(agg @ f2out_w[i] + f2out_b[i])
    gemm128_kernel<1><<<nRowTiles, 256, 0, stream>>>(
        aggH, w_f2o, f2out_b + i * 128, nullptr, nullptr, tH);
    // x = x + t @ dense_w[i] + dense_b[i]   (also refresh f16 copy of x)
    gemm128_kernel<2><<<nRowTiles, 256, 0, stream>>>(
        tH, w_den, dense_b + i * 128, xF, xF, xH);
  }
  (void)hipMemcpyAsync(d_out, xF, (size_t)BA * 128 * 4,
                       hipMemcpyDeviceToDevice, stream);
}
